// AuTx1D_31868657336498
// MI455X (gfx1250) — matrix operand order repaired and checked by reading, not compiled, not run on hardware
//
#include <hip/hip_runtime.h>
#include <hip/hip_bf16.h>
#include <math.h>

// ---------------------------------------------------------------------------
// Types
// ---------------------------------------------------------------------------
typedef __bf16 bf16_t;
typedef __attribute__((ext_vector_type(8)))  bf16_t v8bf;
typedef __attribute__((ext_vector_type(16))) bf16_t v16bf;
typedef __attribute__((ext_vector_type(8)))  float  v8f;

// ---------------------------------------------------------------------------
// Problem constants (from reference)
// ---------------------------------------------------------------------------
#define BB    64
#define SS    256
#define SEQ   257          // S + 1 (agg token)
#define DD    768
#define FF    3072
#define HH    12
#define DKK   64
#define TT    131072
#define KC    512          // conv window
#define M0    (BB*SS)      // 16384  (conv GEMM rows)
#define M1    (BB*SEQ)     // 16448  (transformer rows)  -- divisible by 64
#define QKVN  (3*DD)       // 2304   -- divisible by 256

// ---------------------------------------------------------------------------
// WMMA helpers
// ---------------------------------------------------------------------------
__device__ __forceinline__ v16bf ld16(const bf16_t* p0, const bf16_t* p1) {
  v8bf lo = *(const v8bf*)p0;
  v8bf hi = *(const v8bf*)p1;
  v16bf r;
#pragma unroll
  for (int i = 0; i < 8; ++i) { r[i] = lo[i]; r[i + 8] = hi[i]; }
  return r;
}

__device__ __forceinline__ v8f wmma_bf16(v16bf a, v16bf b, v8f c) {
  // D = A(16x32 bf16) * B(32x16 bf16) + C(16x16 f32)
  return __builtin_amdgcn_wmma_f32_16x16x32_bf16(false, a, false, b,
                                                 (short)0, c, false, false);
}

__device__ __forceinline__ float gelu_tanh(float x) {
  const float c = 0.7978845608028654f;  // sqrt(2/pi)
  return 0.5f * x * (1.0f + tanhf(c * (x + 0.044715f * x * x * x)));
}

// Async global -> LDS copy of 16 bytes per lane (CDNA5, tracked by ASYNCcnt).
// lds_off: byte offset inside the workgroup LDS allocation (per lane).
// base:    uniform 64-bit global base (SGPR pair), voff: per-lane byte offset.
__device__ __forceinline__ void async_ld_b128(unsigned lds_off, const void* base,
                                              unsigned voff) {
  asm volatile("global_load_async_to_lds_b128 %0, %1, %2"
               :: "v"(lds_off), "v"(voff), "s"(base)
               : "memory");
}
__device__ __forceinline__ void wait_async0() {
  asm volatile("s_wait_asynccnt 0" ::: "memory");
}

// ---------------------------------------------------------------------------
// Elementwise conversion f32 -> bf16
// ---------------------------------------------------------------------------
__global__ void k_cvt_bf16(const float* __restrict__ in, bf16_t* __restrict__ out, int n) {
  int i = blockIdx.x * 256 + threadIdx.x;
  if (i < n) out[i] = (bf16_t)in[i];
}

// ---------------------------------------------------------------------------
// Transpose+convert weights: W[K][N] (f32, row-major) -> WT[N][K] (bf16)
// ---------------------------------------------------------------------------
__global__ void k_transpose_bf16(const float* __restrict__ W, bf16_t* __restrict__ WT,
                                 int K, int N) {
  int i = blockIdx.x * 256 + threadIdx.x;
  if (i >= K * N) return;
  int k = i / N, n = i % N;
  WT[(size_t)n * K + k] = (bf16_t)W[i];
}

// Concatenate three bias vectors of length DD into one of length 3*DD.
__global__ void k_concat3(const float* __restrict__ a, const float* __restrict__ b,
                          const float* __restrict__ c, float* __restrict__ o) {
  int i = blockIdx.x * 256 + threadIdx.x;
  if (i >= QKVN) return;
  o[i] = (i < DD) ? a[i] : (i < 2 * DD) ? b[i - DD] : c[i - 2 * DD];
}

// ---------------------------------------------------------------------------
// WMMA GEMM with double-buffered async LDS staging.
//   C[M,N] = A[M,K](bf16) * BT[N,K]^T(bf16) + bias[N]
// 256 threads = 8 waves; wave tile = 32x64 (2x4 16x16 WMMA tiles);
// block tile = 64 x 256, k-step 32.  Requires M%64==0, N%256==0, K%32==0.
// ACT: 0 = none, 1 = tanh-gelu.
// ---------------------------------------------------------------------------
struct GemmSmem {
  bf16_t As[2][64][32];    //  8 KB
  bf16_t Bs[2][256][32];   // 32 KB
};

template <int ACT>
__global__ __launch_bounds__(256)
void k_gemm(const bf16_t* __restrict__ A, const bf16_t* __restrict__ BT,
            const float* __restrict__ bias, float* __restrict__ outF,
            bf16_t* __restrict__ outB, int M, int N, int K) {
  __shared__ GemmSmem sm;
  const int tid  = threadIdx.x;
  const int lane = tid & 31;
  const int wave = tid >> 5;
  const int wm   = wave & 1;          // 0..1  (32-row half)
  const int wn   = wave >> 1;         // 0..3  (64-col group)
  const int m0b  = blockIdx.y * 64;
  const int n0b  = blockIdx.x * 256;
  const int lh   = lane & 15;
  const int a0   = (lane < 16) ? 0 : 8;    // A fragment K sub-offset
  const int b0   = (lane < 16) ? 0 : 8;    // B fragment K sub-offset

  // Staging thread mapping: A tile 64x32 -> 1 b128/thread, B tile 256x32 -> 4.
  const int ar = tid >> 2, aseg = tid & 3;
  const size_t abase = ((size_t)(m0b + ar) * K + aseg * 8) * 2;
  const size_t bbase = ((size_t)(n0b + tid) * K) * 2;
  const char* smbase = (const char*)&sm;

  v8f acc[2][4];
#pragma unroll
  for (int i = 0; i < 2; ++i)
#pragma unroll
    for (int j = 0; j < 4; ++j)
#pragma unroll
      for (int r = 0; r < 8; ++r) acc[i][j][r] = 0.0f;

  const int nk = K >> 5;
  auto issue = [&](int t, int buf) {
    unsigned koff = (unsigned)t * 64u;  // 32 elems * 2B
    async_ld_b128((unsigned)((const char*)&sm.As[buf][ar][aseg * 8] - smbase),
                  A, (unsigned)abase + koff);
#pragma unroll
    for (int c = 0; c < 4; ++c)
      async_ld_b128((unsigned)((const char*)&sm.Bs[buf][tid][c * 8] - smbase),
                    BT, (unsigned)bbase + koff + (unsigned)c * 16u);
  };

  issue(0, 0);
  wait_async0();
  __syncthreads();

  for (int t = 0; t < nk; ++t) {
    int buf = t & 1;
    if (t + 1 < nk) issue(t + 1, buf ^ 1);

    v16bf af[2], bfr[4];
#pragma unroll
    for (int i = 0; i < 2; ++i) {
      const bf16_t* p = &sm.As[buf][wm * 32 + i * 16 + lh][0];
      af[i] = ld16(p + a0, p + a0 + 16);
    }
#pragma unroll
    for (int j = 0; j < 4; ++j) {
      const bf16_t* p = &sm.Bs[buf][wn * 64 + j * 16 + lh][0];
      bfr[j] = ld16(p + b0, p + b0 + 16);
    }
#pragma unroll
    for (int i = 0; i < 2; ++i)
#pragma unroll
      for (int j = 0; j < 4; ++j)
        acc[i][j] = wmma_bf16(af[i], bfr[j], acc[i][j]);

    if (t + 1 < nk) wait_async0();
    __syncthreads();
  }

  // Epilogue: C layout -> vgpr r, lanes<16 row=r, lanes>=16 row=r+8, col=lane%16
#pragma unroll
  for (int i = 0; i < 2; ++i) {
#pragma unroll
    for (int r = 0; r < 8; ++r) {
      int row = m0b + wm * 32 + i * 16 + ((lane < 16) ? r : r + 8);
      size_t rb = (size_t)row * N;
#pragma unroll
      for (int j = 0; j < 4; ++j) {
        int col = n0b + wn * 64 + j * 16 + lh;
        float v = acc[i][j][r] + bias[col];
        if (ACT == 1) v = gelu_tanh(v);
        if (outF) outF[rb + col] = v;
        if (outB) outB[rb + col] = (bf16_t)v;
      }
    }
  }
}

// ---------------------------------------------------------------------------
// Masking + pos-emb + agg token: build h (f32 + bf16), write mask_pos.
// One block per (b, srow) with srow in [0,257).
// ---------------------------------------------------------------------------
__global__ __launch_bounds__(256)
void k_build_h(const float* __restrict__ emb, const float* __restrict__ rnd3,
               const int* __restrict__ perm, const float* __restrict__ pos_emb,
               const float* __restrict__ mask_tok, const float* __restrict__ agg_tok,
               float* __restrict__ h, bf16_t* __restrict__ hb,
               float* __restrict__ mask_pos) {
  int blk = blockIdx.x;                // b*257 + srow
  int b = blk / SEQ, srow = blk % SEQ;
  int tid = threadIdx.x;
  size_t ho = (size_t)blk * DD;
  if (srow == 0) {
    for (int d = tid; d < DD; d += 256) {
      float v = agg_tok[d];
      h[ho + d] = v; hb[ho + d] = (bf16_t)v;
    }
    return;
  }
  int s = srow - 1;
  int idx = b * SS + s;
  float r0 = rnd3[idx * 3 + 0], r1 = rnd3[idx * 3 + 1], r2 = rnd3[idx * 3 + 2];
  bool sel = (r0 <= 0.2f);
  float m  = (sel && r1 <= 0.8f) ? 1.0f : 0.0f;
  float rn = (sel && r1 > 0.8f && r2 <= 0.5f) ? 1.0f : 0.0f;
  float nt = (sel && r1 > 0.8f && r2 >  0.5f) ? 1.0f : 0.0f;
  if (tid == 0) mask_pos[idx] = m + rn + nt;
  int p = perm[idx];
  const float* xr = emb + (size_t)idx * DD;
  const float* sr = emb + (size_t)p   * DD;
  float keep = 1.0f - m - rn;
  for (int d = tid; d < DD; d += 256) {
    float v = xr[d] * keep + mask_tok[d] * m + sr[d] * rn + pos_emb[s * DD + d];
    h[ho + d] = v; hb[ho + d] = (bf16_t)v;
  }
}

// ---------------------------------------------------------------------------
// Flash attention (two-pass softmax), one block per (b, h): 8 waves.
// QKV is the fused projection output: row stride 2304, Q at +0, K at +768,
// V at +1536 (each [257][12][64] within its slice).
// Pass 1: row max only (one butterfly per row per q-tile).
// Pass 2: exact exp(s - max), P*V accumulate, single sum butterfly at end.
// ---------------------------------------------------------------------------
#define VT_STRIDE 288
__global__ __launch_bounds__(256)
void k_attention(const bf16_t* __restrict__ QKV, bf16_t* __restrict__ O) {
  __shared__ __align__(16) bf16_t Vt[DKK][VT_STRIDE];   // 36,864 B
  __shared__ __align__(16) bf16_t Pb[8][16][32];        //  8,192 B

  const int b  = blockIdx.x / HH;
  const int hh = blockIdx.x % HH;
  const int tid  = threadIdx.x;
  const int lane = tid & 31;
  const int wave = tid >> 5;
  const int lh   = lane & 15;
  const bool lo16 = (lane < 16);
  const int a0 = lo16 ? 0 : 8;
  const int b0 = lo16 ? 0 : 8;
  const float scale = 0.125f;  // 1/sqrt(64)

  // Stage V transposed (pad keys >= 257 with zeros).
  for (int idx = tid; idx < DKK * VT_STRIDE; idx += 256) {
    int feat = idx / VT_STRIDE, key = idx % VT_STRIDE;
    bf16_t val = (bf16_t)0.0f;
    if (key < SEQ)
      val = QKV[(size_t)(b * SEQ + key) * QKVN + 2 * DD + hh * DKK + feat];
    Vt[feat][key] = val;
  }
  __syncthreads();

  for (int qt = wave; qt < 17; qt += 8) {     // 17 q-tiles of 16 (257 padded)
    int qrow = qt * 16 + lh; if (qrow > SEQ - 1) qrow = SEQ - 1;
    const bf16_t* qp = QKV + (size_t)(b * SEQ + qrow) * QKVN + hh * DKK;
    v16bf aq0 = ld16(qp + 0  + a0, qp + 0  + a0 + 16);
    v16bf aq1 = ld16(qp + 32 + a0, qp + 32 + a0 + 16);

    // Score tiles for key-pair kp: s0 = cols base..base+15, s1 = +16..+31.
    auto scores = [&](int kp, v8f& s0, v8f& s1) {
      int keybase = kp * 32;
      int key0 = keybase + lh;      if (key0 > SEQ - 1) key0 = SEQ - 1;
      int key1 = keybase + 16 + lh; if (key1 > SEQ - 1) key1 = SEQ - 1;
      const bf16_t* kr0 = QKV + (size_t)(b * SEQ + key0) * QKVN + DD + hh * DKK;
      const bf16_t* kr1 = QKV + (size_t)(b * SEQ + key1) * QKVN + DD + hh * DKK;
#pragma unroll
      for (int r = 0; r < 8; ++r) { s0[r] = 0.0f; s1[r] = 0.0f; }
      s0 = wmma_bf16(aq0, ld16(kr0 + 0  + b0, kr0 + 0  + b0 + 16), s0);
      s0 = wmma_bf16(aq1, ld16(kr0 + 32 + b0, kr0 + 32 + b0 + 16), s0);
      s1 = wmma_bf16(aq0, ld16(kr1 + 0  + b0, kr1 + 0  + b0 + 16), s1);
      s1 = wmma_bf16(aq1, ld16(kr1 + 32 + b0, kr1 + 32 + b0 + 16), s1);
    };

    // ---- Pass 1: row maxima -------------------------------------------
    float mpart[8];
#pragma unroll
    for (int r = 0; r < 8; ++r) mpart[r] = -INFINITY;

    for (int kp = 0; kp < 9; ++kp) {
      v8f s0, s1;
      scores(kp, s0, s1);
      int col0 = kp * 32 + lh, col1 = kp * 32 + 16 + lh;
      bool bad0 = (col0 > SEQ - 1), bad1 = (col1 > SEQ - 1);
#pragma unroll
      for (int r = 0; r < 8; ++r) {
        float v0 = bad0 ? -3.0e38f : s0[r] * scale;
        float v1 = bad1 ? -3.0e38f : s1[r] * scale;
        mpart[r] = fmaxf(mpart[r], fmaxf(v0, v1));
      }
    }
    float mrow[8];
#pragma unroll
    for (int r = 0; r < 8; ++r) {
      float t = mpart[r];
      t = fmaxf(t, __shfl_xor(t, 1, 16));
      t = fmaxf(t, __shfl_xor(t, 2, 16));
      t = fmaxf(t, __shfl_xor(t, 4, 16));
      t = fmaxf(t, __shfl_xor(t, 8, 16));
      mrow[r] = t;
    }

    // ---- Pass 2: exp, P*V accumulate, row sums ------------------------
    float lpart[8];
    v8f acc[4];
#pragma unroll
    for (int r = 0; r < 8; ++r) lpart[r] = 0.0f;
#pragma unroll
    for (int f = 0; f < 4; ++f)
#pragma unroll
      for (int r = 0; r < 8; ++r) acc[f][r] = 0.0f;

    for (int kp = 0; kp < 9; ++kp) {
      int keybase = kp * 32;
      v8f s0, s1;
      scores(kp, s0, s1);
      int col0 = keybase + lh, col1 = keybase + 16 + lh;
      bool bad0 = (col0 > SEQ - 1), bad1 = (col1 > SEQ - 1);
#pragma unroll
      for (int r = 0; r < 8; ++r) {
        float v0 = bad0 ? -3.0e38f : s0[r] * scale;
        float v1 = bad1 ? -3.0e38f : s1[r] * scale;
        float p0 = expf(v0 - mrow[r]);
        float p1 = expf(v1 - mrow[r]);
        lpart[r] += p0 + p1;
        int prow = lo16 ? r : r + 8;
        Pb[wave][prow][lh]      = (bf16_t)p0;
        Pb[wave][prow][16 + lh] = (bf16_t)p1;
      }
      // P (16x32) as A fragment from LDS, V fragments from transposed LDS.
      v16bf ap = ld16(&Pb[wave][lh][a0], &Pb[wave][lh][a0 + 16]);
#pragma unroll
      for (int f = 0; f < 4; ++f) {
        int fc = f * 16 + lh;
        v16bf bv = ld16(&Vt[fc][keybase + b0], &Vt[fc][keybase + b0 + 16]);
        acc[f] = wmma_bf16(ap, bv, acc[f]);
      }
    }

    float lrow[8];
#pragma unroll
    for (int r = 0; r < 8; ++r) {
      float t = lpart[r];
      t += __shfl_xor(t, 1, 16);
      t += __shfl_xor(t, 2, 16);
      t += __shfl_xor(t, 4, 16);
      t += __shfl_xor(t, 8, 16);
      lrow[r] = t;
    }

    // Epilogue: divide by row sum, store bf16.
#pragma unroll
    for (int r = 0; r < 8; ++r) {
      int row = qt * 16 + (lo16 ? r : r + 8);
      if (row < SEQ) {
        float inv = 1.0f / lrow[r];
#pragma unroll
        for (int f = 0; f < 4; ++f) {
          O[(size_t)(b * SEQ + row) * DD + hh * DKK + f * 16 + lh] =
              (bf16_t)(acc[f][r] * inv);
        }
      }
    }
  }
}

// ---------------------------------------------------------------------------
// Residual + LayerNorm (fp32), one block (256 thr) per row of 768.
// Writes h (f32) and hb (bf16).
// ---------------------------------------------------------------------------
__global__ __launch_bounds__(256)
void k_ln_residual(const float* __restrict__ h, const float* __restrict__ pj,
                   const float* __restrict__ g, const float* __restrict__ be,
                   float* __restrict__ ho, bf16_t* __restrict__ hb) {
  __shared__ float red[8];
  int row = blockIdx.x;
  int tid = threadIdx.x;
  int lane = tid & 31, wave = tid >> 5;
  size_t rb = (size_t)row * DD;

  float v[3]; float s = 0.0f;
#pragma unroll
  for (int j = 0; j < 3; ++j) {
    int d = tid + j * 256;
    v[j] = h[rb + d] + pj[rb + d];
    s += v[j];
  }
  s += __shfl_xor(s, 16); s += __shfl_xor(s, 8); s += __shfl_xor(s, 4);
  s += __shfl_xor(s, 2);  s += __shfl_xor(s, 1);
  if (lane == 0) red[wave] = s;
  __syncthreads();
  float tot = 0.0f;
#pragma unroll
  for (int i = 0; i < 8; ++i) tot += red[i];
  float mu = tot * (1.0f / DD);
  __syncthreads();

  float vs = 0.0f;
#pragma unroll
  for (int j = 0; j < 3; ++j) { float d = v[j] - mu; vs += d * d; }
  vs += __shfl_xor(vs, 16); vs += __shfl_xor(vs, 8); vs += __shfl_xor(vs, 4);
  vs += __shfl_xor(vs, 2);  vs += __shfl_xor(vs, 1);
  if (lane == 0) red[wave] = vs;
  __syncthreads();
  float vtot = 0.0f;
#pragma unroll
  for (int i = 0; i < 8; ++i) vtot += red[i];
  float rstd = rsqrtf(vtot * (1.0f / DD) + 1e-6f);

#pragma unroll
  for (int j = 0; j < 3; ++j) {
    int d = tid + j * 256;
    float o = (v[j] - mu) * rstd * g[d] + be[d];
    ho[rb + d] = o;
    hb[rb + d] = (bf16_t)o;
  }
}

// ---------------------------------------------------------------------------
// Final gather: aggregated = h[:,0,:], predictions = h[:,1:,:]
// ---------------------------------------------------------------------------
__global__ void k_gather(const float* __restrict__ h, float* __restrict__ agg,
                         float* __restrict__ pred) {
  size_t i = (size_t)blockIdx.x * 256 + threadIdx.x;
  if (i >= (size_t)M1 * DD) return;
  int row = (int)(i / DD), d = (int)(i % DD);
  int b = row / SEQ, r = row % SEQ;
  float v = h[i];
  if (r == 0) agg[(size_t)b * DD + d] = v;
  else pred[(size_t)(b * SS + (r - 1)) * DD + d] = v;
}

// ---------------------------------------------------------------------------
// Host driver
// ---------------------------------------------------------------------------
static inline size_t alignUp(size_t x) { return (x + 255) & ~(size_t)255; }

extern "C" void kernel_launch(void* const* d_in, const int* in_sizes, int n_in,
                              void* d_out, int out_size, void* d_ws, size_t ws_size,
                              hipStream_t stream) {
  (void)in_sizes; (void)n_in; (void)out_size; (void)ws_size;

  const float* inputs   = (const float*)d_in[0];
  const float* rnd3     = (const float*)d_in[1];
  const int*   perm     = (const int*)d_in[2];
  const float* conv_w   = (const float*)d_in[3];
  const float* conv_b   = (const float*)d_in[4];
  const float* pos_emb  = (const float*)d_in[5];
  const float* mask_tok = (const float*)d_in[6];
  const float* agg_tok  = (const float*)d_in[7];
  const float* wq = (const float*)d_in[8];   const float* bq = (const float*)d_in[9];
  const float* wk = (const float*)d_in[10];  const float* bk = (const float*)d_in[11];
  const float* wv = (const float*)d_in[12];  const float* bv = (const float*)d_in[13];
  const float* wo = (const float*)d_in[14];  const float* bo = (const float*)d_in[15];
  const float* ln1_g = (const float*)d_in[16]; const float* ln1_b = (const float*)d_in[17];
  const float* w1 = (const float*)d_in[18];  const float* b1 = (const float*)d_in[19];
  const float* w2 = (const float*)d_in[20];  const float* b2 = (const float*)d_in[21];
  const float* ln2_g = (const float*)d_in[22]; const float* ln2_b = (const float*)d_in[23];

  // Output regions (concatenated flat, return order).
  float* out      = (float*)d_out;
  float* out_agg  = out;
  float* out_pred = out_agg  + (size_t)BB * DD;
  float* out_mask = out_pred + (size_t)BB * SS * DD;
  float* out_emb  = out_mask + (size_t)BB * SS;

  // Workspace carve.
  uint8_t* wsb = (uint8_t*)d_ws;
  size_t off = 0;
  auto carve = [&](size_t bytes) -> void* {
    void* p = wsb + off; off = alignUp(off + bytes); return p;
  };
  bf16_t* in_bf = (bf16_t*)carve((size_t)BB * TT * 2);
  bf16_t* convT = (bf16_t*)carve((size_t)KC * DD * 2);
  bf16_t *qkvT[2], *woT[2], *w1T[2], *w2T[2];
  float  *bqkv[2];
  for (int l = 0; l < 2; ++l) {
    qkvT[l] = (bf16_t*)carve((size_t)QKVN * DD * 2);  // [2304][768]
    woT[l]  = (bf16_t*)carve((size_t)DD * DD * 2);
    w1T[l]  = (bf16_t*)carve((size_t)DD * FF * 2);
    w2T[l]  = (bf16_t*)carve((size_t)FF * DD * 2);
    bqkv[l] = (float*)carve((size_t)QKVN * 4);
  }
  float*  h    = (float*)carve((size_t)M1 * DD * 4);
  bf16_t* hb   = (bf16_t*)carve((size_t)M1 * DD * 2);
  bf16_t* qkvb = (bf16_t*)carve((size_t)M1 * QKVN * 2);
  bf16_t* ob   = (bf16_t*)carve((size_t)M1 * DD * 2);
  float*  proj = (float*)carve((size_t)M1 * DD * 4);
  bf16_t* fb   = (bf16_t*)carve((size_t)M1 * FF * 2);

  // 1) inputs -> bf16 patch matrix [16384][512] (pure reshape).
  {
    int n = BB * TT;
    k_cvt_bf16<<<(n + 255) / 256, 256, 0, stream>>>(inputs, in_bf, n);
  }
  // 2) transpose/convert weights (QKV fused along N).
  {
    int n = KC * DD;
    k_transpose_bf16<<<(n + 255) / 256, 256, 0, stream>>>(conv_w, convT, KC, DD);
  }
  for (int l = 0; l < 2; ++l) {
    int nd = DD * DD, n1 = DD * FF;
    k_transpose_bf16<<<(nd + 255) / 256, 256, 0, stream>>>(
        wq + (size_t)l * nd, qkvT[l] + 0 * (size_t)DD * DD, DD, DD);
    k_transpose_bf16<<<(nd + 255) / 256, 256, 0, stream>>>(
        wk + (size_t)l * nd, qkvT[l] + 1 * (size_t)DD * DD, DD, DD);
    k_transpose_bf16<<<(nd + 255) / 256, 256, 0, stream>>>(
        wv + (size_t)l * nd, qkvT[l] + 2 * (size_t)DD * DD, DD, DD);
    k_transpose_bf16<<<(nd + 255) / 256, 256, 0, stream>>>(
        wo + (size_t)l * nd, woT[l], DD, DD);
    k_transpose_bf16<<<(n1 + 255) / 256, 256, 0, stream>>>(
        w1 + (size_t)l * n1, w1T[l], DD, FF);
    k_transpose_bf16<<<(n1 + 255) / 256, 256, 0, stream>>>(
        w2 + (size_t)l * n1, w2T[l], FF, DD);
    k_concat3<<<(QKVN + 255) / 256, 256, 0, stream>>>(
        bq + (size_t)l * DD, bk + (size_t)l * DD, bv + (size_t)l * DD, bqkv[l]);
  }
  // 3) Conv as GEMM -> embeddings (f32, directly in d_out).
  k_gemm<0><<<dim3(DD / 256, M0 / 64), 256, 0, stream>>>(
      in_bf, convT, conv_b, out_emb, (bf16_t*)nullptr, M0, DD, KC);
  // 4) Mask + pos-emb + agg token -> h / hb / mask_pos.
  k_build_h<<<M1, 256, 0, stream>>>(out_emb, rnd3, perm, pos_emb, mask_tok,
                                    agg_tok, h, hb, out_mask);
  // 5) Transformer layers.
  for (int l = 0; l < 2; ++l) {
    // Fused QKV projection: one pass over hb.
    k_gemm<0><<<dim3(QKVN / 256, M1 / 64), 256, 0, stream>>>(
        hb, qkvT[l], bqkv[l], (float*)nullptr, qkvb, M1, QKVN, DD);
    k_attention<<<BB * HH, 256, 0, stream>>>(qkvb, ob);
    k_gemm<0><<<dim3(DD / 256, M1 / 64), 256, 0, stream>>>(
        ob, woT[l], bo + (size_t)l * DD, proj, (bf16_t*)nullptr, M1, DD, DD);
    k_ln_residual<<<M1, 256, 0, stream>>>(h, proj, ln1_g + (size_t)l * DD,
                                          ln1_b + (size_t)l * DD, h, hb);
    k_gemm<1><<<dim3(FF / 256, M1 / 64), 256, 0, stream>>>(
        hb, w1T[l], b1 + (size_t)l * FF, (float*)nullptr, fb, M1, FF, DD);
    k_gemm<0><<<dim3(DD / 256, M1 / 64), 256, 0, stream>>>(
        fb, w2T[l], b2 + (size_t)l * DD, proj, (bf16_t*)nullptr, M1, DD, FF);
    k_ln_residual<<<M1, 256, 0, stream>>>(h, proj, ln2_g + (size_t)l * DD,
                                          ln2_b + (size_t)l * DD, h, hb);
  }
  // 6) Final gather.
  {
    size_t n = (size_t)M1 * DD;
    k_gather<<<(unsigned)((n + 255) / 256), 256, 0, stream>>>(h, out_agg, out_pred);
  }
}
